// Attention_89764816486949
// MI455X (gfx1250) — matrix operand order repaired and checked by reading, not compiled, not run on hardware
//
#include <hip/hip_runtime.h>

#define B_      2
#define T_      2048
#define D_      2048
#define NH_     16
#define H_      128
#define WINDOW_ 1024
#define SOFT_CAP_ 50.0f
#define K_MASK_ (-2.3819763e+38f)

typedef __bf16 bf16_t;
typedef __attribute__((ext_vector_type(16))) __bf16 v16bf;
typedef __attribute__((ext_vector_type(8)))  __bf16 v8bf;
typedef __attribute__((ext_vector_type(8)))  float  v8f;
typedef __attribute__((ext_vector_type(4)))  unsigned int u32x4;
typedef __attribute__((ext_vector_type(8)))  unsigned int u32x8;

__device__ inline bf16_t f2bf(float f) { return (bf16_t)f; }

__device__ inline v8f vzero8() {
  v8f z;
#pragma unroll
  for (int i = 0; i < 8; ++i) z[i] = 0.0f;
  return z;
}

__device__ inline v8f wmma_bf16(v16bf a, v16bf b, v8f c) {
  // D = A(16x32 bf16) * B(32x16 bf16) + C(16x16 f32)
  return __builtin_amdgcn_wmma_f32_16x16x32_bf16(
      /*neg_a=*/false, a, /*neg_b=*/false, b,
      /*c_mod=*/(short)0, c, /*reuse_a=*/false, /*reuse_b=*/false);
}

// A-matrix fragment (16x32 bf16). Lane l holds row M=l&15.
// kb = (lane<16 ? 0 : 8). Elements e -> K=2*kb+e (0..15 contiguous).
__device__ inline v16bf load_a_frag(const bf16_t* row, int kb) {
  v8bf lo = *(const v8bf*)(row + 2 * kb);
  v8bf hi = *(const v8bf*)(row + 2 * kb + 8);
  v16bf r;
#pragma unroll
  for (int i = 0; i < 8; ++i) { r[i] = lo[i]; r[i + 8] = hi[i]; }
  return r;
}

// B-matrix fragment (32x16 bf16). Lane l holds col N=l&15.
// kb = (lane<16 ? 0 : 16). Elements e -> K=kb+e (contiguous 16 bf16 = 32B load).
__device__ inline v16bf load_b_frag(const bf16_t* col, int kb) {
  return *(const v16bf*)(col + kb);
}

// ---------------------------------------------------------------------------
// Kernel 1: QKV projection GEMM (bf16 WMMA, f32 accum) + fused RoPE epilogue.
// grid = (M/128 = 32, 3*N = 48), block = 256 (8 waves); each wave: 16x128 strip.
// q,k stored [B,N,T,H] bf16 ; v stored [B,N,H,T] bf16 (transposed for PV).
// ---------------------------------------------------------------------------
__global__ __launch_bounds__(256)
void qkv_rope_kernel(const float* __restrict__ x, const float* __restrict__ wqkv,
                     bf16_t* __restrict__ qo, bf16_t* __restrict__ ko,
                     bf16_t* __restrict__ vo) {
  __shared__ bf16_t As[128][48];   // 128 rows x 32 K (stride 48 keeps alignment)
  __shared__ bf16_t Bt[128][48];   // transposed: [h][K]
  const int tid  = threadIdx.x;
  const int wave = tid >> 5, lane = tid & 31;
  const int lr = lane & 15, hi = lane >> 4;
  const int blk_m = blockIdx.x;          // 0..31
  const int sn    = blockIdx.y;          // 0..47
  const int s = sn >> 4, n = sn & 15;
  const float* wslice = wqkv + (size_t)sn * D_ * H_;
  const float* xbase  = x + (size_t)blk_m * 128 * D_;

  v8f acc[8];
#pragma unroll
  for (int t = 0; t < 8; ++t) acc[t] = vzero8();

  for (int k0 = 0; k0 < D_; k0 += 32) {
    __syncthreads();
    // stage A: x[128][32] f32 -> bf16
#pragma unroll
    for (int it = 0; it < 4; ++it) {
      int row = it * 32 + (tid >> 3);
      int c4  = (tid & 7) << 2;
      const float4 av = *(const float4*)(xbase + (size_t)row * D_ + k0 + c4);
      As[row][c4 + 0] = f2bf(av.x); As[row][c4 + 1] = f2bf(av.y);
      As[row][c4 + 2] = f2bf(av.z); As[row][c4 + 3] = f2bf(av.w);
    }
    // stage B transposed: w[k][h] f32 -> Bt[h][k] bf16
#pragma unroll
    for (int it = 0; it < 4; ++it) {
      int kk = it * 8 + (tid >> 5);
      int h4 = (tid & 31) << 2;
      const float4 bv = *(const float4*)(wslice + (size_t)(k0 + kk) * H_ + h4);
      Bt[h4 + 0][kk] = f2bf(bv.x); Bt[h4 + 1][kk] = f2bf(bv.y);
      Bt[h4 + 2][kk] = f2bf(bv.z); Bt[h4 + 3][kk] = f2bf(bv.w);
    }
    if (k0 + 32 < D_) {
      __builtin_prefetch(xbase + (size_t)(tid >> 3) * D_ + k0 + 32, 0, 0);
      __builtin_prefetch(wslice + (size_t)(k0 + 32 + (tid >> 5)) * H_, 0, 0);
    }
    __syncthreads();

    v16bf af = load_a_frag(&As[wave * 16 + lr][0], hi * 8);
#pragma unroll
    for (int t = 0; t < 8; ++t) {
      v16bf bfr = load_b_frag(&Bt[t * 16 + lr][0], hi * 16);
      acc[t] = wmma_bf16(af, bfr, acc[t]);
    }
  }

  // Epilogue: RoPE for q/k (pairs are acc[t] / acc[t+4], same lane), scale q.
  const int m0 = blk_m * 128 + wave * 16;
  if (s < 2) {
    const float qscale = (s == 0) ? 0.08838834764831845f : 1.0f;  // H^-0.5
    bf16_t* out = (s == 0) ? qo : ko;
#pragma unroll
    for (int r = 0; r < 8; ++r) {
      int m = m0 + r + 8 * hi;
      int b = m >> 11;            // / T_
      int pos = m & (T_ - 1);
      size_t base = (((size_t)(b * NH_ + n) * T_) + pos) * H_;
#pragma unroll
      for (int t = 0; t < 4; ++t) {
        int h = t * 16 + lr;      // h in [0,64)
        float ang = (float)pos * __expf(-(float)h * 0.14391156831212787f); // ln(1e4)/64
        float sn_, cs_;
        __sincosf(ang, &sn_, &cs_);
        float fi = acc[t][r], se = acc[t + 4][r];
        out[base + h]      = f2bf((fi * cs_ - se * sn_) * qscale);
        out[base + h + 64] = f2bf((se * cs_ + fi * sn_) * qscale);
      }
    }
  } else {
#pragma unroll
    for (int r = 0; r < 8; ++r) {
      int m = m0 + r + 8 * hi;
      int b = m >> 11;
      int pos = m & (T_ - 1);
#pragma unroll
      for (int t = 0; t < 8; ++t) {
        int h = t * 16 + lr;
        vo[(((size_t)(b * NH_ + n) * H_) + h) * T_ + pos] = f2bf(acc[t][r]);
      }
    }
  }
}

// ---------------------------------------------------------------------------
// Kernel 2: sliding-window flash attention with tanh soft-cap.
// grid = B*N*(T/64), block = 128 (4 waves); wave handles 16 queries, 32 keys/step.
// enc written as [B*T, N*H] bf16 (A operand of out-proj).
// ---------------------------------------------------------------------------
__global__ __launch_bounds__(128)
void attn_kernel(const bf16_t* __restrict__ q, const bf16_t* __restrict__ k,
                 const bf16_t* __restrict__ v, bf16_t* __restrict__ enc) {
  __shared__ bf16_t Pb[4][16][32];   // per-wave P transpose buffer
  const int tid  = threadIdx.x;
  const int wave = tid >> 5, lane = tid & 31;
  const int lr = lane & 15, hi = lane >> 4;
  const int nqb = T_ / 64;
  const int bx = blockIdx.x;
  const int qblk = bx % nqb;
  const int n = (bx / nqb) % NH_;
  const int b = bx / (nqb * NH_);
  const int q_base = qblk * 64 + wave * 16;
  const size_t bn = (size_t)(b * NH_ + n);
  const bf16_t* qb = q + (bn * T_ + q_base) * H_;
  const bf16_t* kb = k + bn * T_ * H_;
  const bf16_t* vb = v + bn * H_ * T_;

  // Q A-fragments: 4 chunks of 16x32 over H=128
  v16bf qa[4];
#pragma unroll
  for (int c = 0; c < 4; ++c)
    qa[c] = load_a_frag(qb + (size_t)lr * H_ + c * 32, hi * 8);

  v8f acc[8];
  float mrow[8], lrow[8];
#pragma unroll
  for (int t = 0; t < 8; ++t) acc[t] = vzero8();
#pragma unroll
  for (int r = 0; r < 8; ++r) { mrow[r] = -3.0e38f; lrow[r] = 0.0f; }

  int klo = q_base - WINDOW_ + 1;
  if (klo < 0) klo = 0;
  klo &= ~31;
  const int khi = q_base + 15;

  for (int ks = klo; ks <= khi; ks += 32) {
    // S = Q * K^T : two 16x16 tiles (keys ks..ks+15, ks+16..ks+31)
    v8f s0 = vzero8(), s1 = vzero8();
#pragma unroll
    for (int c = 0; c < 4; ++c) {
      v16bf kf0 = load_b_frag(kb + (size_t)(ks + lr) * H_ + c * 32, hi * 16);
      s0 = wmma_bf16(qa[c], kf0, s0);
      v16bf kf1 = load_b_frag(kb + (size_t)(ks + 16 + lr) * H_ + c * 32, hi * 16);
      s1 = wmma_bf16(qa[c], kf1, s1);
    }
    // soft-cap, mask, online softmax (row = r + 8*hi, col spread over 16 lanes)
    float p0[8], p1[8];
#pragma unroll
    for (int r = 0; r < 8; ++r) {
      int qrow = q_base + r + 8 * hi;
      float x0 = tanhf(s0[r] * (1.0f / SOFT_CAP_)) * SOFT_CAP_;
      float x1 = tanhf(s1[r] * (1.0f / SOFT_CAP_)) * SOFT_CAP_;
      int k0i = ks + lr, k1i = ks + 16 + lr;
      x0 = ((k0i <= qrow) && (k0i > qrow - WINDOW_)) ? x0 : K_MASK_;
      x1 = ((k1i <= qrow) && (k1i > qrow - WINDOW_)) ? x1 : K_MASK_;
      float mx = fmaxf(x0, x1);
#pragma unroll
      for (int off = 1; off < 16; off <<= 1) mx = fmaxf(mx, __shfl_xor(mx, off));
      float mnew = fmaxf(mrow[r], mx);
      float corr = __expf(mrow[r] - mnew);
      float e0 = __expf(x0 - mnew);
      float e1 = __expf(x1 - mnew);
      float rs = e0 + e1;
#pragma unroll
      for (int off = 1; off < 16; off <<= 1) rs += __shfl_xor(rs, off);
      lrow[r] = lrow[r] * corr + rs;
      mrow[r] = mnew;
      p0[r] = e0; p1[r] = e1;
#pragma unroll
      for (int t = 0; t < 8; ++t) acc[t][r] *= corr;
    }
    // transpose P (C layout) -> A layout via per-wave LDS
#pragma unroll
    for (int r = 0; r < 8; ++r) {
      int row = r + 8 * hi;
      Pb[wave][row][lr]      = f2bf(p0[r]);
      Pb[wave][row][16 + lr] = f2bf(p1[r]);
    }
    v16bf pa = load_a_frag(&Pb[wave][lr][0], hi * 8);
    // O += P * V  (V stored [B,N,H,T]: contiguous keys per h-column)
#pragma unroll
    for (int t = 0; t < 8; ++t) {
      v16bf vf = load_b_frag(vb + (size_t)(t * 16 + lr) * T_ + ks, hi * 16);
      acc[t] = wmma_bf16(pa, vf, acc[t]);
    }
  }

  // normalize and store encoded [B*T, N*H] bf16
#pragma unroll
  for (int r = 0; r < 8; ++r) {
    int qrow = q_base + r + 8 * hi;
    float inv = 1.0f / lrow[r];
    size_t base = ((size_t)(b * T_ + qrow)) * (NH_ * H_) + n * H_;
#pragma unroll
    for (int t = 0; t < 8; ++t)
      enc[base + t * 16 + lr] = f2bf(acc[t][r] * inv);
  }
}

// ---------------------------------------------------------------------------
// Kernel 3: output projection GEMM: enc[4096,2048]bf16 x w_out[2048,2048]f32 -> f32
// grid = (32, 16), block = 256 (8 waves x 16x128 strips).
// A-tile (bf16, no conversion needed) is staged by the Tensor Data Mover:
// one 2D tensor_load_to_lds per K-step, with hardware LDS padding reproducing
// the 48-element LDS row stride (16 DW row + 8 DW pad = 96B).
// ---------------------------------------------------------------------------
__global__ __launch_bounds__(256)
void outproj_kernel(const bf16_t* __restrict__ enc, const float* __restrict__ wout,
                    float* __restrict__ out) {
  __shared__ bf16_t As[128][48];
  __shared__ bf16_t Bt[128][48];
  const int tid  = threadIdx.x;
  const int wave = tid >> 5, lane = tid & 31;
  const int lr = lane & 15, hi = lane >> 4;
  const int blk_m = blockIdx.x;   // 0..31
  const int blk_n = blockIdx.y;   // 0..15
  const bf16_t* abase = enc + (size_t)blk_m * 128 * D_;
  const float*  bbase = wout + blk_n * 128;

  // --- TDM descriptor group 1 (loop-invariant): 2D tile 32x128 of 2B elems ---
  // data_size=1(2B); pad_enable; pad_interval=16DW(code 3); pad_amount=8DW(code 7)
  // tensor_dim0=2048, tensor_dim1=4096, tile_dim0=32, tile_dim1=128, dim0_stride=2048
  u32x8 g1;
  g1[0] = (1u << 16) | (1u << 20) | (3u << 22) | (7u << 25);
  g1[1] = (2048u << 16);            // tensor_dim0 low16 (bits 63:48)
  g1[2] = (4096u << 16);            // tensor_dim0 hi16=0 | tensor_dim1 low16
  g1[3] = (32u << 16);              // tensor_dim1 hi16=0 | tile_dim0
  g1[4] = 128u;                     // tile_dim1 | tile_dim2=0
  g1[5] = 2048u;                    // tensor_dim0_stride low32
  g1[6] = 0u;                       // stride hi16 | tensor_dim1_stride low16
  g1[7] = 0u;                       // tensor_dim1_stride hi32
  const unsigned int lds_a = (unsigned int)(size_t)(&As[0][0]); // LDS byte offset
  const unsigned long long ga0 = (unsigned long long)(size_t)abase;

  v8f acc[8];
#pragma unroll
  for (int t = 0; t < 8; ++t) acc[t] = vzero8();

  for (int k0 = 0; k0 < D_; k0 += 32) {
    __syncthreads();   // previous-iteration readers done before DMA/stores
    if (wave == 0) {
      unsigned long long ga = ga0 + (unsigned long long)k0 * 2u;
      u32x4 g0;
      g0[0] = 1u;                                   // count=1, user mode
      g0[1] = lds_a;                                // lds_addr
      g0[2] = (unsigned int)ga;                     // global_addr[31:0]
      g0[3] = ((unsigned int)(ga >> 32) & 0x01FFFFFFu) | (2u << 30); // addr[56:32]|type=2
      asm volatile("tensor_load_to_lds %0, %1" :: "s"(g0), "s"(g1) : "memory");
    }
    // stage B transposed, f32 -> bf16 (overlaps with TDM transfer)
#pragma unroll
    for (int it = 0; it < 4; ++it) {
      int kk = it * 8 + (tid >> 5);
      int h4 = (tid & 31) << 2;
      const float4 bv = *(const float4*)(bbase + (size_t)(k0 + kk) * D_ + h4);
      Bt[h4 + 0][kk] = f2bf(bv.x); Bt[h4 + 1][kk] = f2bf(bv.y);
      Bt[h4 + 2][kk] = f2bf(bv.z); Bt[h4 + 3][kk] = f2bf(bv.w);
    }
    if (wave == 0) __builtin_amdgcn_s_wait_tensorcnt(0);
    __syncthreads();

    v16bf af = load_a_frag(&As[wave * 16 + lr][0], hi * 8);
#pragma unroll
    for (int t = 0; t < 8; ++t) {
      v16bf bfr = load_b_frag(&Bt[t * 16 + lr][0], hi * 16);
      acc[t] = wmma_bf16(af, bfr, acc[t]);
    }
  }

  const int m0 = blk_m * 128 + wave * 16;
#pragma unroll
  for (int r = 0; r < 8; ++r) {
    int m = m0 + r + 8 * hi;
    float* orow = out + (size_t)m * D_ + blk_n * 128;
#pragma unroll
    for (int t = 0; t < 8; ++t) orow[t * 16 + lr] = acc[t][r];
  }
}

// ---------------------------------------------------------------------------
extern "C" void kernel_launch(void* const* d_in, const int* in_sizes, int n_in,
                              void* d_out, int out_size, void* d_ws, size_t ws_size,
                              hipStream_t stream) {
  (void)in_sizes; (void)n_in; (void)out_size; (void)ws_size;
  const float* x    = (const float*)d_in[0];
  const float* wqkv = (const float*)d_in[1];
  const float* wout = (const float*)d_in[2];
  // d_in[3] = segment_pos (arange, recomputed in-kernel), d_in[4] = causal mask (baked in)

  const size_t per = (size_t)B_ * NH_ * T_ * H_;   // 8.39M elements
  bf16_t* qw  = (bf16_t*)d_ws;
  bf16_t* kw  = qw + per;
  bf16_t* vw  = kw + per;
  bf16_t* encw = vw + per;                          // [B*T, N*H]
  float* out = (float*)d_out;

  qkv_rope_kernel<<<dim3(32, 48), dim3(256), 0, stream>>>(x, wqkv, qw, kw, vw);
  attn_kernel<<<dim3(B_ * NH_ * (T_ / 64)), dim3(128), 0, stream>>>(qw, kw, vw, encw);
  outproj_kernel<<<dim3(32, 16), dim3(256), 0, stream>>>(encw, wout, out);
}
